// Vocoder_89833535963941
// MI455X (gfx1250) — matrix operand order repaired and checked by reading, not compiled, not run on hardware
//
#include <hip/hip_runtime.h>
#include <hip/hip_bf16.h>
#include <math.h>

// ---------------- problem constants (from setup_inputs) ----------------
#define BB    8
#define CM    192
#define TT    2048
#define DD    512
#define HH    1536
#define NFFT  1024
#define HOP   256
#define NF    513                 // n_fft/2+1
#define MROWS (BB*TT)             // 16384
#define KIM   (CM*7)              // 1344 (im2col K)
#define NHEAD (2*NF)              // 1026
#define NFP   544                 // nF padded to mult of 32
#define KA2   (2*NFP)             // 1088 ([Sr|Si] packed K)
#define LOUT  ((TT-1)*HOP)        // 524032 cropped samples per batch

typedef __attribute__((ext_vector_type(16))) _Float16 v16h;
typedef __attribute__((ext_vector_type(8)))  float    v8f;

union FragH { v16h h; unsigned u[8]; };
union AccF  { v8f  v; float    f[8]; };

#define LDSPITCH 40               // 32 halves padded to 40 (kills 4-way bank conflicts)

// 16B-per-lane async global->LDS copy (ASYNCcnt); lds_off = wave-relative LDS byte offset
__device__ __forceinline__ void async_ld16(unsigned lds_off, const void* gaddr)
{
    asm volatile("global_load_async_to_lds_b128 %0, %1, off"
                 :: "v"(lds_off), "v"(gaddr) : "memory");
}
__device__ __forceinline__ void wait_async0()
{
    asm volatile("s_wait_asynccnt 0x0" ::: "memory");
}

// ======================================================================
// Tiled WMMA GEMM: C(MxN) = A(MxK,f16,row-major) * W(NxK,f16,row-major)^T
// 256 threads = 8 waves; macro tile 128x128; waves 4(M)x2(N); wave tile 32x64.
// Double-buffered async global->LDS pipeline, one barrier per K-step.
// EPI: 0 = +bias -> f32 | 1 = +bias,GELU -> f16 | 2 = resid + gamma*(acc+bias) -> f32
//      3 = plain -> f32
// NOTE: tile loads are unguarded in N; caller guarantees W rows up to the next
//       multiple of 128 are readable (workspace padding). Stores guard col < N.
// ======================================================================
template<int EPI>
__global__ __launch_bounds__(256)
void gemm_f16_wmma(const _Float16* __restrict__ A,
                   const _Float16* __restrict__ W,
                   const float* __restrict__ bias,
                   const float* __restrict__ gamma,
                   const float* __restrict__ resid,
                   float* __restrict__ outF,
                   _Float16* __restrict__ outH,
                   int N, int K)
{
    __shared__ _Float16 lA[2][128 * LDSPITCH];
    __shared__ _Float16 lW[2][128 * LDSPITCH];

    const int tid  = threadIdx.x;
    const int bM   = blockIdx.y, bN = blockIdx.x;
    const int wid  = tid >> 5,  lane = tid & 31;
    const int half = lane >> 4, l16  = lane & 15;
    const int wM   = wid >> 1,  wN   = wid & 1;

    // ---- per-thread load slots: tile is 128 rows x 4 chunks of 8 halves; 2 slots each ----
    int r0 = tid >> 1;                        // slot i: idx = tid + i*256
    int c0 = (tid & 1) * 2;                   // chunk pairs: (c0, c0+?) -> use idx mapping below
    (void)r0; (void)c0;
    int ra[2], ca[2];
#pragma unroll
    for (int i = 0; i < 2; ++i) { int idx = tid + i * 256; ra[i] = idx >> 2; ca[i] = idx & 3; }

    const _Float16* gA[2];
    const _Float16* gW[2];
    unsigned laOff[2], lwOff[2];
    const unsigned bufStride = 128 * LDSPITCH * 2;   // bytes per buffer
    const unsigned baseA = (unsigned)(size_t)&lA[0][0];
    const unsigned baseW = (unsigned)(size_t)&lW[0][0];
#pragma unroll
    for (int i = 0; i < 2; ++i) {
        gA[i] = A + (size_t)(bM * 128 + ra[i]) * K + ca[i] * 8;
        gW[i] = W + (size_t)(bN * 128 + ra[i]) * K + ca[i] * 8;
        laOff[i] = baseA + (unsigned)(ra[i] * LDSPITCH + ca[i] * 8) * 2;
        lwOff[i] = baseW + (unsigned)(ra[i] * LDSPITCH + ca[i] * 8) * 2;
    }

    AccF acc[2][4];
#pragma unroll
    for (int i = 0; i < 2; ++i)
#pragma unroll
        for (int j = 0; j < 4; ++j)
#pragma unroll
            for (int r = 0; r < 8; ++r) acc[i][j].f[r] = 0.f;

    const int nK = K >> 5;

    // prologue: issue tile 0 into buffer 0
#pragma unroll
    for (int i = 0; i < 2; ++i) {
        async_ld16(laOff[i], gA[i]);
        async_ld16(lwOff[i], gW[i]);
    }

    for (int ks = 0; ks < nK; ++ks) {
        const int cur = ks & 1;
        wait_async0();          // own loads for buf[cur] have landed in LDS
        __syncthreads();        // all waves loaded buf[cur]; all waves done computing buf[cur^1]

        if (ks + 1 < nK) {      // issue next tile into the buffer just freed
            const unsigned bofs = (unsigned)((cur ^ 1) * bufStride);
#pragma unroll
            for (int i = 0; i < 2; ++i) {
                gA[i] += 32; gW[i] += 32;
                async_ld16(laOff[i] + bofs, gA[i]);
                async_ld16(lwOff[i] + bofs, gW[i]);
            }
        }

        const _Float16* sA = lA[cur];
        const _Float16* sW = lW[cur];
        FragH af[2], bf[4];
#pragma unroll
        for (int ms = 0; ms < 2; ++ms) {            // A 16x32 f16 layout (ISA 7.12.2)
            int m = wM * 32 + ms * 16 + l16;
#pragma unroll
            for (int j = 0; j < 8; ++j) {
                int k = ((j >> 2) << 4) + half * 8 + ((j & 3) << 1);
                af[ms].u[j] = *(const unsigned*)(sA + m * LDSPITCH + k);
            }
        }
#pragma unroll
        for (int ns = 0; ns < 4; ++ns) {            // B 32x16: lane=n, vgpr j holds K=(j>>2)*16+half*8+2(j&3)
            int n = wN * 64 + ns * 16 + l16;
#pragma unroll
            for (int j = 0; j < 8; ++j) {
                int k = ((j >> 2) << 4) + half * 8 + ((j & 3) << 1);
                bf[ns].u[j] = *(const unsigned*)(sW + n * LDSPITCH + k);
            }
        }
#pragma unroll
        for (int ms = 0; ms < 2; ++ms)
#pragma unroll
            for (int ns = 0; ns < 4; ++ns)
                acc[ms][ns].v = __builtin_amdgcn_wmma_f32_16x16x32_f16(
                    false, af[ms].h, false, bf[ns].h,
                    (short)0, acc[ms][ns].v, false, false);
    }

    // epilogue: C/D 16x16 f32 layout: vgpr r -> (M = half*8 + r, N = l16)
#pragma unroll
    for (int ms = 0; ms < 2; ++ms)
#pragma unroll
        for (int ns = 0; ns < 4; ++ns)
#pragma unroll
            for (int r = 0; r < 8; ++r) {
                int row = bM * 128 + wM * 32 + ms * 16 + half * 8 + r;
                int col = bN * 128 + wN * 64 + ns * 16 + l16;
                if (col >= N) continue;
                size_t idx = (size_t)row * N + col;
                float v = acc[ms][ns].f[r];
                if (EPI == 0) {
                    outF[idx] = v + bias[col];
                } else if (EPI == 1) {
                    float t = v + bias[col];
                    outH[idx] = (_Float16)(0.5f * t * (1.f + erff(t * 0.70710678118f)));
                } else if (EPI == 2) {
                    outF[idx] = resid[idx] + gamma[col] * (v + bias[col]);
                } else {
                    outF[idx] = v;
                }
            }
}

// ---------------- block-wide (128 thr) sum/sumsq reduction ----------------
__device__ inline void block_reduce2(float& s, float& q, float* sh)
{
#pragma unroll
    for (int off = 16; off >= 1; off >>= 1) {
        s += __shfl_xor(s, off);
        q += __shfl_xor(q, off);
    }
    int wid = threadIdx.x >> 5, lane = threadIdx.x & 31;
    if (lane == 0) { sh[wid * 2] = s; sh[wid * 2 + 1] = q; }
    __syncthreads();
    if (threadIdx.x == 0) {
        float S = 0.f, Q = 0.f;
        for (int i = 0; i < 4; ++i) { S += sh[2 * i]; Q += sh[2 * i + 1]; }
        sh[0] = S; sh[1] = Q;
    }
    __syncthreads();
    s = sh[0]; q = sh[1];
}

// ---------------- LayerNorm over D=512 per (b,t) row ----------------
template<bool OUT16>
__global__ __launch_bounds__(128)
void ln_kernel(const float* __restrict__ in, const float* __restrict__ w,
               const float* __restrict__ b, float* __restrict__ outF,
               _Float16* __restrict__ outH)
{
    __shared__ float sh[8];
    size_t row = blockIdx.x;
    int d = threadIdx.x * 4;
    float4 x = *(const float4*)(in + row * DD + d);
    float s = x.x + x.y + x.z + x.w;
    float q = x.x * x.x + x.y * x.y + x.z * x.z + x.w * x.w;
    block_reduce2(s, q, sh);
    float mean = s * (1.f / DD);
    float var  = q * (1.f / DD) - mean * mean;
    float rs   = rsqrtf(var + 1e-5f);
    float o0 = (x.x - mean) * rs * w[d + 0] + b[d + 0];
    float o1 = (x.y - mean) * rs * w[d + 1] + b[d + 1];
    float o2 = (x.z - mean) * rs * w[d + 2] + b[d + 2];
    float o3 = (x.w - mean) * rs * w[d + 3] + b[d + 3];
    if (OUT16) {
        outH[row * DD + d + 0] = (_Float16)o0;
        outH[row * DD + d + 1] = (_Float16)o1;
        outH[row * DD + d + 2] = (_Float16)o2;
        outH[row * DD + d + 3] = (_Float16)o3;
    } else {
        float4 o = {o0, o1, o2, o3};
        *(float4*)(outF + row * DD + d) = o;
    }
}

// ------------- depthwise conv7 (+bias) + LayerNorm -> f16 -------------
__global__ __launch_bounds__(128)
void dwconv_ln_kernel(const float* __restrict__ x, const float* __restrict__ dww,
                      const float* __restrict__ dwb, const float* __restrict__ nw,
                      const float* __restrict__ nb, _Float16* __restrict__ y)
{
    __shared__ float sh[8];
    int bt = blockIdx.x;
    int t  = bt & (TT - 1);
    int d  = threadIdx.x * 4;
    float a0 = 0.f, a1 = 0.f, a2 = 0.f, a3 = 0.f;
#pragma unroll
    for (int k = 0; k < 7; ++k) {
        int t2 = t + k - 3;
        if (t2 < 0 || t2 >= TT) continue;
        float4 xv = *(const float4*)(x + (size_t)(bt + k - 3) * DD + d);
        a0 += xv.x * dww[(d + 0) * 7 + k];
        a1 += xv.y * dww[(d + 1) * 7 + k];
        a2 += xv.z * dww[(d + 2) * 7 + k];
        a3 += xv.w * dww[(d + 3) * 7 + k];
    }
    a0 += dwb[d + 0]; a1 += dwb[d + 1]; a2 += dwb[d + 2]; a3 += dwb[d + 3];
    float s = a0 + a1 + a2 + a3;
    float q = a0 * a0 + a1 * a1 + a2 * a2 + a3 * a3;
    block_reduce2(s, q, sh);
    float mean = s * (1.f / DD);
    float var  = q * (1.f / DD) - mean * mean;
    float rs   = rsqrtf(var + 1e-5f);
    size_t base = (size_t)bt * DD + d;
    y[base + 0] = (_Float16)((a0 - mean) * rs * nw[d + 0] + nb[d + 0]);
    y[base + 1] = (_Float16)((a1 - mean) * rs * nw[d + 1] + nb[d + 1]);
    y[base + 2] = (_Float16)((a2 - mean) * rs * nw[d + 2] + nb[d + 2]);
    y[base + 3] = (_Float16)((a3 - mean) * rs * nw[d + 3] + nb[d + 3]);
}

// ---------------- im2col for the embed conv (K=192*7) ----------------
__global__ void im2col_kernel(const float* __restrict__ mel, _Float16* __restrict__ A0)
{
    size_t idx = (size_t)blockIdx.x * blockDim.x + threadIdx.x;
    if (idx >= (size_t)MROWS * KIM) return;
    int f  = (int)(idx % KIM);
    int bt = (int)(idx / KIM);
    int c = f / 7, k = f % 7;
    int b = bt / TT, t = bt % TT;
    int t2 = t + k - 3;
    float v = 0.f;
    if (t2 >= 0 && t2 < TT) v = mel[((size_t)b * CM + c) * TT + t2];
    A0[idx] = (_Float16)v;
}

// ------------- iDFT basis [cos ; -sin] * scale, f16, K-packed -------------
__global__ void basis_kernel(_Float16* __restrict__ W2)
{
    int idx = blockIdx.x * blockDim.x + threadIdx.x;
    if (idx >= NFFT * KA2) return;
    int n = idx / KA2, f2 = idx % KA2;
    float v = 0.f;
    int f = (f2 < NFP) ? f2 : (f2 - NFP);
    if (f < NF) {
        int r = (n * f) & (NFFT - 1);                     // exact phase reduction
        float ang   = (float)r * (6.283185307179586f / NFFT);
        float scale = (f == 0 || f == NF - 1) ? (1.f / NFFT) : (2.f / NFFT);
        v = (f2 < NFP) ? cosf(ang) * scale : -sinf(ang) * scale;
    }
    W2[idx] = (_Float16)v;
}

// ------- head output -> packed [S_real | S_imag] f16 GEMM operand -------
__global__ void headpost_kernel(const float* __restrict__ hout, _Float16* __restrict__ A2)
{
    int idx = blockIdx.x * blockDim.x + threadIdx.x;
    if (idx >= MROWS * NFP) return;
    int t = idx / NFP, f = idx % NFP;
    float sr = 0.f, si = 0.f;
    if (f < NF) {
        float m  = fminf(expf(hout[(size_t)t * NHEAD + f]), 100.f);
        float ph = hout[(size_t)t * NHEAD + NF + f];
        sr = m * cosf(ph);
        si = m * sinf(ph);
    }
    A2[(size_t)t * KA2 + f]       = (_Float16)sr;
    A2[(size_t)t * KA2 + NFP + f] = (_Float16)si;
}

// ------- overlap-add (deterministic 4-tap gather) + window-env norm -------
__global__ void ola_kernel(const float* __restrict__ frames, const float* __restrict__ window,
                           float* __restrict__ out)
{
    size_t idx = (size_t)blockIdx.x * blockDim.x + threadIdx.x;
    if (idx >= (size_t)BB * LOUT) return;
    int b  = (int)(idx / LOUT);
    int lo = (int)(idx % LOUT);
    int l  = lo + NFFT / 2;           // undo the center crop
    float a = 0.f, e = 0.f;
    int tmax = l / HOP;
#pragma unroll
    for (int dt = 3; dt >= 0; --dt) {
        int t = tmax - dt;
        if (t < 0 || t >= TT) continue;
        int j = l - t * HOP;
        if (j < 0 || j >= NFFT) continue;
        float wv = window[j];
        a += frames[((size_t)b * TT + t) * NFFT + j] * wv;
        e += wv * wv;
    }
    out[idx] = a / (e + 1e-11f);
}

__global__ void f32_to_f16_kernel(const float* __restrict__ in, _Float16* __restrict__ out, size_t n)
{
    size_t i = (size_t)blockIdx.x * blockDim.x + threadIdx.x;
    if (i < n) out[i] = (_Float16)in[i];
}

// ======================================================================
extern "C" void kernel_launch(void* const* d_in, const int* in_sizes, int n_in,
                              void* d_out, int out_size, void* d_ws, size_t ws_size,
                              hipStream_t stream)
{
    const float* mel     = (const float*)d_in[0];
    const float* embed_w = (const float*)d_in[1];
    const float* embed_b = (const float*)d_in[2];
    const float* norm_w  = (const float*)d_in[3];
    const float* norm_b  = (const float*)d_in[4];
    const float* bdw_w   = (const float*)d_in[5];
    const float* bdw_b   = (const float*)d_in[6];
    const float* bn_w    = (const float*)d_in[7];
    const float* bn_b    = (const float*)d_in[8];
    const float* bpw1_w  = (const float*)d_in[9];
    const float* bpw1_b  = (const float*)d_in[10];
    const float* bpw2_w  = (const float*)d_in[11];
    const float* bpw2_b  = (const float*)d_in[12];
    const float* bgamma  = (const float*)d_in[13];
    const float* fnorm_w = (const float*)d_in[14];
    const float* fnorm_b = (const float*)d_in[15];
    const float* head_w  = (const float*)d_in[16];
    const float* head_b  = (const float*)d_in[17];
    const float* window  = (const float*)d_in[18];
    (void)in_sizes; (void)n_in; (void)out_size; (void)ws_size;

    // ---- workspace carve (time-multiplexed) ----
    char* ws = (char*)d_ws;
    size_t off = 0;
    auto carve = [&](size_t bytes) -> char* {
        char* p = ws + off; off += (bytes + 255) & ~(size_t)255; return p;
    };
    size_t szIm2col = (size_t)MROWS * KIM * 2;          // 44.0 MB
    size_t szFrames = (size_t)MROWS * NFFT * 4;         // 67.1 MB
    size_t szXtmp   = (size_t)MROWS * DD * 4;           // 33.6 MB
    size_t szA2     = (size_t)MROWS * KA2 * 2;          // 35.7 MB
    size_t szZ      = (size_t)MROWS * HH * 2;           // 50.3 MB
    size_t szHout   = (size_t)MROWS * NHEAD * 4;        // 67.2 MB
    char* buf0 = carve(szIm2col > szFrames ? szIm2col : szFrames); // im2col / frames
    char* buf1 = carve(szXtmp > szA2 ? szXtmp : szA2);             // embed-out / [Sr|Si]
    char* buf2 = carve(szXtmp);                                    // residual stream x
    char* buf3 = carve((size_t)MROWS * DD * 2);                    // y16 / h16
    char* buf4 = carve(szZ > szHout ? szZ : szHout);               // z16 / head-out
    // +32KB tail pad: head GEMM reads W rows up to n=1151 (N padded to 128) unguarded
    _Float16* wf16  = (_Float16*)carve(13796352ull * 2 + 32768);   // converted weights
    _Float16* basis = (_Float16*)carve((size_t)NFFT * KA2 * 2);

    _Float16* w16_embed = wf16;
    _Float16* w16_pw1   = w16_embed + (size_t)DD * KIM;            // 8 x (HH x DD)
    _Float16* w16_pw2   = w16_pw1 + 8ull * HH * DD;                // 8 x (DD x HH)
    _Float16* w16_head  = w16_pw2 + 8ull * DD * HH;                // NHEAD x DD

    // ---- convert weights to f16 ----
    auto cvt = [&](const float* s, _Float16* d, size_t n) {
        f32_to_f16_kernel<<<dim3((unsigned)((n + 255) / 256)), 256, 0, stream>>>(s, d, n);
    };
    cvt(embed_w, w16_embed, (size_t)DD * KIM);
    cvt(bpw1_w,  w16_pw1,   8ull * HH * DD);
    cvt(bpw2_w,  w16_pw2,   8ull * DD * HH);
    cvt(head_w,  w16_head,  (size_t)NHEAD * DD);

    basis_kernel<<<dim3((NFFT * KA2 + 255) / 256), 256, 0, stream>>>(basis);

    // ---- embed: im2col + GEMM(+bias) + LN -> x ----
    im2col_kernel<<<dim3((unsigned)(((size_t)MROWS * KIM + 255) / 256)), 256, 0, stream>>>(
        mel, (_Float16*)buf0);
    gemm_f16_wmma<0><<<dim3(DD / 128, MROWS / 128), 256, 0, stream>>>(
        (const _Float16*)buf0, w16_embed, embed_b, nullptr, nullptr,
        (float*)buf1, nullptr, DD, KIM);
    ln_kernel<false><<<dim3(MROWS), 128, 0, stream>>>(
        (const float*)buf1, norm_w, norm_b, (float*)buf2, nullptr);

    // ---- 8 ConvNeXt blocks ----
    for (int blk = 0; blk < 8; ++blk) {
        dwconv_ln_kernel<<<dim3(MROWS), 128, 0, stream>>>(
            (const float*)buf2, bdw_w + (size_t)blk * DD * 7, bdw_b + (size_t)blk * DD,
            bn_w + (size_t)blk * DD, bn_b + (size_t)blk * DD, (_Float16*)buf3);
        gemm_f16_wmma<1><<<dim3(HH / 128, MROWS / 128), 256, 0, stream>>>(
            (const _Float16*)buf3, w16_pw1 + (size_t)blk * HH * DD,
            bpw1_b + (size_t)blk * HH, nullptr, nullptr,
            nullptr, (_Float16*)buf4, HH, DD);
        gemm_f16_wmma<2><<<dim3(DD / 128, MROWS / 128), 256, 0, stream>>>(
            (const _Float16*)buf4, w16_pw2 + (size_t)blk * DD * HH,
            bpw2_b + (size_t)blk * DD, bgamma + (size_t)blk * DD, (const float*)buf2,
            (float*)buf2, nullptr, DD, HH);
    }

    // ---- final LN -> f16, head GEMM, spectrum, iSTFT GEMM, OLA ----
    ln_kernel<true><<<dim3(MROWS), 128, 0, stream>>>(
        (const float*)buf2, fnorm_w, fnorm_b, nullptr, (_Float16*)buf3);
    gemm_f16_wmma<0><<<dim3((NHEAD + 127) / 128, MROWS / 128), 256, 0, stream>>>(
        (const _Float16*)buf3, w16_head, head_b, nullptr, nullptr,
        (float*)buf4, nullptr, NHEAD, DD);
    headpost_kernel<<<dim3((MROWS * NFP + 255) / 256), 256, 0, stream>>>(
        (const float*)buf4, (_Float16*)buf1);
    gemm_f16_wmma<3><<<dim3(NFFT / 128, MROWS / 128), 256, 0, stream>>>(
        (const _Float16*)buf1, basis, nullptr, nullptr, nullptr,
        (float*)buf0, nullptr, NFFT, KA2);
    ola_kernel<<<dim3((unsigned)(((size_t)BB * LOUT + 255) / 256)), 256, 0, stream>>>(
        (const float*)buf0, window, (float*)d_out);
}
